// LambdaLayer_15547781612042
// MI455X (gfx1250) — compile-verified, matrix operand order checked by reading, not run on hardware
//
#include <hip/hip_runtime.h>
#include <hip/hip_bf16.h>
#include <math.h>

// ---------- types ----------
typedef __attribute__((ext_vector_type(16))) __bf16          v16bf;
typedef __attribute__((ext_vector_type(16))) unsigned short  u16x16;
typedef __attribute__((ext_vector_type(8)))  unsigned int    u32x8;
typedef __attribute__((ext_vector_type(8)))  float           v8f;
typedef __attribute__((ext_vector_type(2)))  float           v2f;
typedef __attribute__((ext_vector_type(4)))  unsigned int    v4u;
typedef __attribute__((ext_vector_type(8)))  int             v8i;
typedef __attribute__((ext_vector_type(4)))  int             v4i;

#if defined(__has_builtin)
#  if __has_builtin(__builtin_amdgcn_tensor_load_to_lds)
#    define HAVE_TDM 1
#  else
#    define HAVE_TDM 0
#  endif
#else
#  define HAVE_TDM 0
#endif

// ---------- helpers ----------
__device__ __forceinline__ unsigned short f32_to_bf16(float f) {
  unsigned u = __builtin_bit_cast(unsigned, f);
  u += 0x7FFFu + ((u >> 16) & 1u);          // round-to-nearest-even
  return (unsigned short)(u >> 16);
}

// 16 bf16 fragment from global: halves at p[0..7] and p[16..23]
// (WMMA 16-bit layout: per-lane K = {0..7,16..23} or {8..15,24..31})
__device__ __forceinline__ v16bf frag_gap16(const unsigned short* p) {
  const uint4 a = *(const uint4*)p;
  const uint4 b = *(const uint4*)(p + 16);
  u32x8 u;
  u[0]=a.x; u[1]=a.y; u[2]=a.z; u[3]=a.w;
  u[4]=b.x; u[5]=b.y; u[6]=b.z; u[7]=b.w;
  return __builtin_bit_cast(v16bf, u);
}

// 16 bf16 fragment from 16 contiguous ushorts (LDS, already in lane order)
__device__ __forceinline__ v16bf frag_contig(const unsigned short* p) {
  const uint4 a = *(const uint4*)p;
  const uint4 b = *(const uint4*)(p + 8);
  u32x8 u;
  u[0]=a.x; u[1]=a.y; u[2]=a.z; u[3]=a.w;
  u[4]=b.x; u[5]=b.y; u[6]=b.z; u[7]=b.w;
  return __builtin_bit_cast(v16bf, u);
}

#define MT 128          // m-tile per block in K1/K6
#define NCH 144         // qkv channels
#define MTOT 4096       // h*w
#define BATCH 32

#if HAVE_TDM
// ---- Tensor Data Mover: 2D f32 tile (tile_h rows x tile_w elems) global -> LDS ----
// D# packing per CDNA5 ISA ch.8: Group0 {count,lds_addr,global_addr,type=2},
// Group1 {data_size=4B, tensor_dim0/1, tile_dim0/1, tensor_dim0_stride}.
__device__ __forceinline__ void tdm_load_2d_f32(const float* gptr, unsigned lds_byte,
                                                unsigned tile_w, unsigned tile_h,
                                                unsigned row_stride) {
  const unsigned long long ga = (unsigned long long)(size_t)gptr;
  v4u g0;
  g0[0] = 1u;                                              // count=1 (user descriptor)
  g0[1] = lds_byte;                                        // lds_addr (bytes)
  g0[2] = (unsigned)(ga & 0xFFFFFFFFu);                    // global_addr[31:0]
  g0[3] = (unsigned)((ga >> 32) & 0x01FFFFFFu) | (2u << 30); // addr[56:32] | type=2
  v8i g1;
  g1[0] = (int)(2u << 16);                                 // wgmask=0, data_size=2 (4B)
  g1[1] = (int)(row_stride << 16);                         // tensor_dim0[15:0]
  g1[2] = (int)(tile_h << 16);                             // dim0[31:16]=0, tensor_dim1[15:0]
  g1[3] = (int)(tile_w << 16);                             // dim1[31:16]=0, tile_dim0
  g1[4] = (int)tile_h;                                     // tile_dim1, tile_dim2=0
  g1[5] = (int)row_stride;                                 // tensor_dim0_stride[31:0]
  g1[6] = 0;                                               // stride[47:32], dim1_stride lo
  g1[7] = 0;
  v4i gz = {};                                             // groups 2/3 unused (2D)
#if defined(__clang_major__) && (__clang_major__ >= 23)
  v8i gz8 = {};
  __builtin_amdgcn_tensor_load_to_lds(g0, g1, gz, gz, gz8, 0);
#else
  __builtin_amdgcn_tensor_load_to_lds(g0, g1, gz, gz, 0);
#endif
}
#endif // HAVE_TDM

// ---------- K1: qkv GEMM  out[b,o,m] = sum_c x[b,c,m] * W[o,c] ----------
// grid 1024 = 32 b * 32 mtiles, block 256 (8 waves)
// x tile staged by TDM (double-buffered, overlapped with WMMA via TENSORcnt)
__global__ __launch_bounds__(256)
void k_qkv_gemm(const float* __restrict__ x, const float* __restrict__ W,
                float* __restrict__ qkv) {
  __shared__ unsigned short wfrag[9][32][16];  // A fragments, current K-step: 9216 B
  __shared__ float xs[2][32][MT];              // double-buffered x stage:   32768 B
  const int tid  = threadIdx.x;
  const int lane = tid & 31, wave = tid >> 5;
  const int b  = blockIdx.x >> 5;
  const int m0 = (blockIdx.x & 31) * MT;
  const int hi = (lane >= 16) ? 8 : 0;

  v8f acc[9] = {};

#if HAVE_TDM
  if (tid < 32)  // wave 0 only: TDM issues regardless of EXEC
    tdm_load_2d_f32(x + ((size_t)(b * 256)) * MTOT + m0,
                    (unsigned)(size_t)&xs[0][0][0], MT, 32, MTOT);
#endif

  for (int ks = 0; ks < 8; ++ks) {
    const int cur = ks & 1;
    __syncthreads();
    // stage W fragments for this K-step (bf16, A-matrix lane layout)
    for (int idx = tid; idx < 9 * 32; idx += 256) {
      const int ot = idx >> 5, l = idx & 31;
      const float* wr = W + (ot * 16 + (l & 15)) * 256 + ks * 32 + ((l >= 16) ? 8 : 0);
      if (ks < 7) __builtin_prefetch(wr + 32, 0, 3);   // next K-step of W
      #pragma unroll
      for (int i = 0; i < 16; ++i)
        wfrag[ot][l][i] = f32_to_bf16(wr[(i < 8) ? i : (i + 8)]);
    }
#if HAVE_TDM
    if (tid < 32) {
      if (ks < 7) {   // kick next tile, then wait for the current one (in-order)
        tdm_load_2d_f32(x + ((size_t)(b * 256 + (ks + 1) * 32)) * MTOT + m0,
                        (unsigned)(size_t)&xs[cur ^ 1][0][0], MT, 32, MTOT);
        __builtin_amdgcn_s_wait_tensorcnt(1);
      } else {
        __builtin_amdgcn_s_wait_tensorcnt(0);
      }
    }
#else
    for (int s = tid; s < (32 * MT / 4); s += 256) {
      const int r  = s >> 5;
      const int c4 = (s & 31) * 4;
      *(float4*)&xs[cur][r][c4] =
          *(const float4*)(x + ((size_t)(b * 256 + ks * 32 + r)) * MTOT + m0 + c4);
    }
#endif
    __syncthreads();

    // B fragment (K=32 x N=16) for this wave's m-subtile, from LDS
    u16x16 bt;
    const int n = wave * 16 + (lane & 15);
    #pragma unroll
    for (int i = 0; i < 16; ++i) {
      const int kk = ((i < 8) ? i : (i + 8)) + hi;
      bt[i] = f32_to_bf16(xs[cur][kk][n]);
    }
    const v16bf bfrag = __builtin_bit_cast(v16bf, bt);

    #pragma unroll
    for (int ot = 0; ot < 9; ++ot) {
      const v16bf afrag = frag_contig(&wfrag[ot][lane][0]);
      acc[ot] = __builtin_amdgcn_wmma_f32_16x16x32_bf16(
          false, afrag, false, bfrag, (short)0, acc[ot], false, false);
    }
  }

  const int m = m0 + wave * 16 + (lane & 15);
  #pragma unroll
  for (int ot = 0; ot < 9; ++ot)
    #pragma unroll
    for (int r = 0; r < 8; ++r) {
      const int o = ot * 16 + r + hi;
      qkv[((size_t)(b * NCH + o)) * MTOT + m] = acc[ot][r];
    }
}

// ---------- K2: batchnorm stats -> fused (scale, shift) ----------
// grid 128 (64 q-ch + 64 v-ch), block 256
__global__ __launch_bounds__(256)
void k_bn_stats(const float* __restrict__ qkv,
                const float* __restrict__ gq, const float* __restrict__ bq,
                const float* __restrict__ gv, const float* __restrict__ bv,
                float* __restrict__ stats) {
  __shared__ float s0[256], s1[256];
  const int ch = blockIdx.x;
  const int o  = (ch < 64) ? ch : (ch + 16);   // v channels start at 80
  float sum = 0.f, sq = 0.f;
  for (int i = threadIdx.x; i < BATCH * MTOT; i += 256) {
    const float v = qkv[((size_t)((i >> 12) * NCH + o)) * MTOT + (i & 4095)];
    sum += v; sq += v * v;
  }
  s0[threadIdx.x] = sum; s1[threadIdx.x] = sq;
  __syncthreads();
  for (int off = 128; off > 0; off >>= 1) {
    if (threadIdx.x < off) {
      s0[threadIdx.x] += s0[threadIdx.x + off];
      s1[threadIdx.x] += s1[threadIdx.x + off];
    }
    __syncthreads();
  }
  if (threadIdx.x == 0) {
    const float inv_n = 1.0f / (float)(BATCH * MTOT);
    const float mean  = s0[0] * inv_n;
    const float var   = fmaxf(s1[0] * inv_n - mean * mean, 0.0f);
    const float g  = (ch < 64) ? gq[ch] : gv[ch - 64];
    const float be = (ch < 64) ? bq[ch] : bv[ch - 64];
    const float scale = g * rsqrtf(var + 1e-5f);
    stats[ch * 2]     = scale;
    stats[ch * 2 + 1] = be - mean * scale;
  }
}

// ---------- K3: softmax over m + positional embedding -> bf16 A matrix ----------
// grid 512 = 32 b * 16 k-rows, block 256
__global__ __launch_bounds__(256)
void k_softmax_addpos(const float* __restrict__ qkv, const float* __restrict__ ff,
                      unsigned short* __restrict__ alam) {
  __shared__ float red[256];
  const int b = blockIdx.x >> 4, kk = blockIdx.x & 15;
  const float* row = qkv + ((size_t)(b * NCH + 64 + kk)) * MTOT;
  float mx = -3.4e38f;
  for (int m = threadIdx.x; m < MTOT; m += 256) mx = fmaxf(mx, row[m]);
  red[threadIdx.x] = mx; __syncthreads();
  for (int off = 128; off > 0; off >>= 1) {
    if (threadIdx.x < off) red[threadIdx.x] = fmaxf(red[threadIdx.x], red[threadIdx.x + off]);
    __syncthreads();
  }
  mx = red[0]; __syncthreads();
  float sum = 0.f;
  for (int m = threadIdx.x; m < MTOT; m += 256) sum += __expf(row[m] - mx);
  red[threadIdx.x] = sum; __syncthreads();
  for (int off = 128; off > 0; off >>= 1) {
    if (threadIdx.x < off) red[threadIdx.x] += red[threadIdx.x + off];
    __syncthreads();
  }
  const float inv = 1.0f / red[0];
  const int jj = kk & 7;
  const float f0 = ff[jj]     * 6.28318530718f;
  const float f1 = ff[8 + jj] * 6.28318530718f;
  for (int m = threadIdx.x; m < MTOT; m += 256) {
    const float gh = -1.0f + 2.0f * (float)(m >> 6) * (1.0f / 63.0f);
    const float gw = -1.0f + 2.0f * (float)(m & 63) * (1.0f / 63.0f);
    const float pr = gh * f0 + gw * f1;
    const float e  = (kk < 8) ? __sinf(pr) : __cosf(pr);
    alam[((size_t)(b * 16 + kk)) * MTOT + m] =
        f32_to_bf16(__expf(row[m] - mx) * inv + e);
  }
}

// ---------- K4: v -> BN -> bf16 B matrix ----------
// grid 8192, block 256; each thread does 4 contiguous m (float4 in, ushort4 out)
__global__ __launch_bounds__(256)
void k_v_bn_bf16(const float* __restrict__ qkv, const float* __restrict__ stats,
                 unsigned short* __restrict__ vbf) {
  const int idx = blockIdx.x * 256 + threadIdx.x;  // 2,097,152 total
  const int m4  = (idx & 1023) * 4;
  const int vch = (idx >> 10) & 63;
  const int b   = idx >> 16;
  const float scale = stats[(64 + vch) * 2];
  const float shift = stats[(64 + vch) * 2 + 1];
  const float4 v = *(const float4*)(qkv + ((size_t)(b * NCH + 80 + vch)) * MTOT + m4);
  ushort4 o;
  o.x = f32_to_bf16(v.x * scale + shift);
  o.y = f32_to_bf16(v.y * scale + shift);
  o.z = f32_to_bf16(v.z * scale + shift);
  o.w = f32_to_bf16(v.w * scale + shift);
  *(ushort4*)(vbf + ((size_t)(b * 64 + vch)) * MTOT + m4) = o;
}

// ---------- K5: lambda[b] (16x64) = Alam(16x4096) @ Vbf(4096x64) ----------
// grid 32 (one per batch), block 256 (8 waves split K)
__global__ __launch_bounds__(256)
void k_lambda_gemm(const unsigned short* __restrict__ alam,
                   const unsigned short* __restrict__ vbf,
                   float* __restrict__ lam) {
  __shared__ float sp[8][4][32][8];                 // 32 KB
  const int b = blockIdx.x;
  const int lane = threadIdx.x & 31, wave = threadIdx.x >> 5;
  const int hi = (lane >= 16) ? 8 : 0;
  v8f acc[4] = {};
  const unsigned short* arow = alam + ((size_t)(b * 16 + (lane & 15))) * MTOT + hi;

  for (int it = 0; it < 16; ++it) {
    const int c0 = wave * 512 + it * 32;
    const v16bf afrag = frag_gap16(arow + c0);
    #pragma unroll
    for (int nt = 0; nt < 4; ++nt) {
      const int col = nt * 16 + (lane & 15);
      const v16bf bfrag = frag_gap16(vbf + ((size_t)(b * 64 + col)) * MTOT + c0 + hi);
      acc[nt] = __builtin_amdgcn_wmma_f32_16x16x32_bf16(
          false, afrag, false, bfrag, (short)0, acc[nt], false, false);
    }
  }
  #pragma unroll
  for (int nt = 0; nt < 4; ++nt)
    #pragma unroll
    for (int r = 0; r < 8; ++r)
      sp[wave][nt][lane][r] = acc[nt][r];
  __syncthreads();
  for (int pos = threadIdx.x; pos < 1024; pos += 256) {
    const int nt = pos >> 8, slot = pos & 255;
    const int ls = slot >> 3, r = slot & 7;
    float s = 0.f;
    #pragma unroll
    for (int w = 0; w < 8; ++w) s += sp[w][nt][ls][r];
    const int k = r + ((ls >= 16) ? 8 : 0);
    const int v = nt * 16 + (ls & 15);
    lam[((size_t)(b * 16 + k)) * 64 + v] = s;
  }
}

// ---------- K6: out[b,head*64+v,m] = sum_k q_bn[b,k*4+head,m] * lam[b,k,v] ----------
// grid 4096 = 32 b * 4 head * 32 mtiles, block 256; f32 WMMA 16x16x4, K=16 in 4 steps
__global__ __launch_bounds__(256)
void k_out_gemm(const float* __restrict__ qkv, const float* __restrict__ stats,
                const float* __restrict__ lam, float* __restrict__ out) {
  const int blk  = blockIdx.x;
  const int mt   = blk & 31, head = (blk >> 5) & 3, b = blk >> 7;
  const int lane = threadIdx.x & 31, wave = threadIdx.x >> 5;
  const int m  = mt * MT + wave * 16 + (lane & 15);
  const int hi = (lane >= 16) ? 8 : 0;

  v2f bf[4];
  #pragma unroll
  for (int s = 0; s < 4; ++s) {               // B = q_bn (K=4 x N=16)
    const int k0 = s * 4 + ((lane < 16) ? 0 : 2);
    const int o0 = k0 * 4 + head, o1 = (k0 + 1) * 4 + head;
    bf[s].x = qkv[((size_t)(b * NCH + o0)) * MTOT + m] * stats[o0 * 2] + stats[o0 * 2 + 1];
    bf[s].y = qkv[((size_t)(b * NCH + o1)) * MTOT + m] * stats[o1 * 2] + stats[o1 * 2 + 1];
  }
  const float* lamb = lam + (size_t)b * 16 * 64;
  #pragma unroll
  for (int vt = 0; vt < 4; ++vt) {
    v8f acc = {};
    const int vv = vt * 16 + (lane & 15);
    #pragma unroll
    for (int s = 0; s < 4; ++s) {             // A = lamT (M=16 x K=4)
      const int k0 = s * 4 + ((lane < 16) ? 0 : 2);
      v2f af;
      af.x = lamb[k0 * 64 + vv];
      af.y = lamb[(k0 + 1) * 64 + vv];
      acc = __builtin_amdgcn_wmma_f32_16x16x4_f32(
          false, af, false, bf[s], (short)0, acc, false, false);
    }
    #pragma unroll
    for (int r = 0; r < 8; ++r) {
      const int v = vt * 16 + r + hi;
      out[((size_t)(b * 256 + head * 64 + v)) * MTOT + m] = acc[r];
    }
  }
}

// ---------- launch ----------
extern "C" void kernel_launch(void* const* d_in, const int* in_sizes, int n_in,
                              void* d_out, int out_size, void* d_ws, size_t ws_size,
                              hipStream_t stream) {
  (void)in_sizes; (void)n_in; (void)out_size; (void)ws_size;
  const float* x  = (const float*)d_in[0];
  const float* W  = (const float*)d_in[1];
  const float* gq = (const float*)d_in[2];
  const float* bq = (const float*)d_in[3];
  const float* gv = (const float*)d_in[4];
  const float* bv = (const float*)d_in[5];
  const float* ff = (const float*)d_in[6];
  float* out = (float*)d_out;

  // workspace layout (bytes): total ~96.6 MB
  char* ws = (char*)d_ws;
  float*          qkv   = (float*)ws;                              // 75,497,472
  unsigned short* alam  = (unsigned short*)(ws + 75497472);        //  4,194,304
  unsigned short* vbf   = (unsigned short*)(ws + 79691776);        // 16,777,216
  float*          stats = (float*)(ws + 96468992);                 //      1,024
  float*          lam   = (float*)(ws + 96470016);                 //    131,072

  k_qkv_gemm      <<<dim3(1024), dim3(256), 0, stream>>>(x, W, qkv);
  k_bn_stats      <<<dim3(128),  dim3(256), 0, stream>>>(qkv, gq, bq, gv, bv, stats);
  k_softmax_addpos<<<dim3(512),  dim3(256), 0, stream>>>(qkv, ff, alam);
  k_v_bn_bf16     <<<dim3(8192), dim3(256), 0, stream>>>(qkv, stats, vbf);
  k_lambda_gemm   <<<dim3(32),   dim3(256), 0, stream>>>(alam, vbf, lam);
  k_out_gemm      <<<dim3(4096), dim3(256), 0, stream>>>(qkv, stats, lam, out);
}
